// AttentionMIL_37950331027980
// MI455X (gfx1250) — matrix operand order repaired and checked by reading, not compiled, not run on hardware
//
#include <hip/hip_runtime.h>
#include <hip/hip_bf16.h>
#include <math.h>

// Problem constants from the reference
#define B_ 16
#define N_ 8192
#define D_ 1024
#define H_ 256

typedef __attribute__((ext_vector_type(16))) _Float16 v16h;
typedef __attribute__((ext_vector_type(8)))  float    v8f;

// ---------------------------------------------------------------------------
// Kernel 0: convert W1 [H, D] fp32 -> f16 (lives in L2; read by every WG)
// ---------------------------------------------------------------------------
__global__ __launch_bounds__(256) void cvt_w1_kernel(const float* __restrict__ W1,
                                                     _Float16* __restrict__ W1h) {
    int g = blockIdx.x * 256 + threadIdx.x;
    if (g < H_ * D_) W1h[g] = (_Float16)W1[g];
}

// ---------------------------------------------------------------------------
// Kernel 1: fused  scores[m] = sum_j tanh( sum_k X[m,k]*W1[j,k] ) * w2[j]
// One workgroup = 8 waves = 128 rows of X. Each wave: 16 rows x 256 cols,
// K-loop in steps of 32 using v_wmma_f32_16x16x32_f16.
// ---------------------------------------------------------------------------
__global__ __launch_bounds__(256) void mlp_scores_kernel(
    const float*    __restrict__ X,
    const _Float16* __restrict__ W1h,
    const float*    __restrict__ W2,
    float*          __restrict__ scores)
{
    const int wave = threadIdx.x >> 5;
    const int lane = threadIdx.x & 31;
    const int half = lane >> 4;     // which 16-lane group
    const int l16  = lane & 15;     // row index (A) / column index (B,D)

    const long rowBase = (long)blockIdx.x * 128 + wave * 16;
    const float* __restrict__ Xrow = X + (rowBase + l16) * (long)D_;

    v8f acc[16];
#pragma unroll
    for (int t = 0; t < 16; ++t) acc[t] = (v8f){0.f,0.f,0.f,0.f,0.f,0.f,0.f,0.f};

    for (int k0 = 0; k0 < D_; k0 += 32) {
        // Prefetch next K-step of this row (speculative; OOB is dropped)
        __builtin_prefetch(Xrow + k0 + 64, 0, 3);

        // ---- A fragment (16x32 f16, contiguous k = 16*half + s order) ----
        // half=0: elems[0..15]=K 0..15
        // half=1: elems[0..15]=K 16..31
        const float4* p1 = (const float4*)(Xrow + k0 + 16 * half);
        const float4* p2 = (const float4*)(Xrow + k0 + 16 * half + 8);
        float4 f0 = p1[0], f1 = p1[1], f2 = p2[0], f3 = p2[1];
        v16h a;
        a[0]  = (_Float16)f0.x; a[1]  = (_Float16)f0.y;
        a[2]  = (_Float16)f0.z; a[3]  = (_Float16)f0.w;
        a[4]  = (_Float16)f1.x; a[5]  = (_Float16)f1.y;
        a[6]  = (_Float16)f1.z; a[7]  = (_Float16)f1.w;
        a[8]  = (_Float16)f2.x; a[9]  = (_Float16)f2.y;
        a[10] = (_Float16)f2.z; a[11] = (_Float16)f2.w;
        a[12] = (_Float16)f3.x; a[13] = (_Float16)f3.y;
        a[14] = (_Float16)f3.z; a[15] = (_Float16)f3.w;

        // ---- 16 column tiles; B[k][n] = W1[c0+n][k]; lane holds 16
        // consecutive K (offset 16*half) of W1 row c0+l16 (32B aligned) ----
#pragma unroll
        for (int t = 0; t < 16; ++t) {
            const v16h b = *(const v16h*)(W1h + (size_t)(t * 16 + l16) * D_
                                               + k0 + 16 * half);
            acc[t] = __builtin_amdgcn_wmma_f32_16x16x32_f16(
                /*neg_a=*/false, a, /*neg_b=*/false, b,
                /*c_mod=*/(short)0, acc[t],
                /*reuse_a=*/false, /*reuse_b=*/false);
        }
    }

    // ---- Epilogue: tanh, dot with w2, reduce across the 16 columns ----
    // D layout: VGPR j / lane  -> element (M = j + 8*half, N = l16)
    float sacc[8];
#pragma unroll
    for (int j = 0; j < 8; ++j) sacc[j] = 0.f;

#pragma unroll
    for (int t = 0; t < 16; ++t) {
        const float w2v = W2[t * 16 + l16];
#pragma unroll
        for (int j = 0; j < 8; ++j)
            sacc[j] = fmaf(tanhf(acc[t][j]), w2v, sacc[j]);
    }
#pragma unroll
    for (int j = 0; j < 8; ++j) {
        sacc[j] += __shfl_xor(sacc[j], 1, 32);
        sacc[j] += __shfl_xor(sacc[j], 2, 32);
        sacc[j] += __shfl_xor(sacc[j], 4, 32);
        sacc[j] += __shfl_xor(sacc[j], 8, 32);   // masks < 16: stays in half-group
    }
    if (l16 == 0) {
        const long r = rowBase + 8 * half;       // lanes 0 and 16 write
#pragma unroll
        for (int j = 0; j < 8; ++j) scores[r + j] = sacc[j];
    }
}

// ---------------------------------------------------------------------------
// Kernel 2: in-place softmax over N per batch (16 blocks)
// ---------------------------------------------------------------------------
__global__ __launch_bounds__(256) void softmax_kernel(float* __restrict__ scores) {
    __shared__ float red[256];
    float* s = scores + (size_t)blockIdx.x * N_;
    const int tid = threadIdx.x;

    float mx = -INFINITY;
    for (int n = tid; n < N_; n += 256) mx = fmaxf(mx, s[n]);
    red[tid] = mx; __syncthreads();
    for (int st = 128; st > 0; st >>= 1) {
        if (tid < st) red[tid] = fmaxf(red[tid], red[tid + st]);
        __syncthreads();
    }
    mx = red[0]; __syncthreads();

    float sum = 0.f;
    for (int n = tid; n < N_; n += 256) {
        float e = __expf(s[n] - mx);
        s[n] = e;
        sum += e;
    }
    red[tid] = sum; __syncthreads();
    for (int st = 128; st > 0; st >>= 1) {
        if (tid < st) red[tid] += red[tid + st];
        __syncthreads();
    }
    const float inv = 1.0f / red[0];
    for (int n = tid; n < N_; n += 256) s[n] *= inv;
}

// ---------------------------------------------------------------------------
// Kernel 3: partial weighted sums  partial[b,nc,d] = sum_{n in chunk} a[b,n]*X[b,n,d]
// 512 blocks: b(16) x dchunk(4) x nchunk(8). Deterministic (no atomics).
// ---------------------------------------------------------------------------
__global__ __launch_bounds__(256) void wsum_partial_kernel(
    const float* __restrict__ X, const float* __restrict__ attn,
    float* __restrict__ partial)
{
    const int blk = blockIdx.x;
    const int b   = blk >> 5;
    const int rem = blk & 31;
    const int dc  = rem >> 3;
    const int nc  = rem & 7;
    const int d   = dc * 256 + threadIdx.x;

    const float* __restrict__ Xb = X    + (size_t)b * N_ * D_;
    const float* __restrict__ ab = attn + (size_t)b * N_;

    float acc = 0.f;
    const int n0 = nc * (N_ / 8);
    for (int n = n0; n < n0 + (N_ / 8); ++n)
        acc = fmaf(ab[n], Xb[(size_t)n * D_ + d], acc);

    partial[((size_t)(b * 8 + nc)) * D_ + d] = acc;
}

// ---------------------------------------------------------------------------
// Kernel 4: reduce the 8 N-chunk partials -> out[b, d]
// ---------------------------------------------------------------------------
__global__ __launch_bounds__(256) void wsum_reduce_kernel(
    const float* __restrict__ partial, float* __restrict__ out)
{
    const int g = blockIdx.x * 256 + threadIdx.x;   // 16384 = 16*1024
    const int b = g >> 10;
    const int d = g & 1023;
    float s = 0.f;
#pragma unroll
    for (int nc = 0; nc < 8; ++nc)
        s += partial[((size_t)(b * 8 + nc)) * D_ + d];
    out[g] = s;
}

// ---------------------------------------------------------------------------
extern "C" void kernel_launch(void* const* d_in, const int* in_sizes, int n_in,
                              void* d_out, int out_size, void* d_ws, size_t ws_size,
                              hipStream_t stream) {
    const float* X  = (const float*)d_in[0];   // [B, N, D]
    const float* W1 = (const float*)d_in[1];   // [H, D]
    const float* W2 = (const float*)d_in[2];   // [1, H]
    float* out = (float*)d_out;                // [B, D]

    // Workspace layout (float slots):
    //   [0      .. 131071] scores / attn (in-place softmax)   512 KB
    //   [131072 .. 262143] W1 as f16 (262144 halfs)           512 KB
    //   [262144 .. 393215] weighted-sum partials [16][8][1024] 512 KB
    float*    ws_f    = (float*)d_ws;
    float*    scores  = ws_f;
    _Float16* W1h     = (_Float16*)(ws_f + 131072);
    float*    partial = ws_f + 262144;

    cvt_w1_kernel     <<<(H_ * D_ + 255) / 256, 256, 0, stream>>>(W1, W1h);
    mlp_scores_kernel <<<(B_ * N_) / 128,       256, 0, stream>>>(X, W1h, W2, scores);
    softmax_kernel    <<<B_,                    256, 0, stream>>>(scores);
    wsum_partial_kernel<<<B_ * 32,              256, 0, stream>>>(X, scores, partial);
    wsum_reduce_kernel <<<(B_ * D_) / 256,      256, 0, stream>>>(partial, out);
}
